// HybridConvNet_80126909874621
// MI455X (gfx1250) — matrix operand order repaired and checked by reading, not compiled, not run on hardware
//
#include <hip/hip_runtime.h>
#include <hip/hip_bf16.h>

// ---------------------------------------------------------------------------
// HGT forward for MI455X (gfx1250):
//   - dense math via v_wmma_f32_16x16x32_bf16 (bf16 in, f32 accum)
//   - LDS staging via global_load_async_to_lds_b128 (ASYNCcnt) + s_wait_asynccnt
//   - activations/weights converted to bf16 once (fused scale+GELU epilogues)
// ---------------------------------------------------------------------------

typedef __attribute__((ext_vector_type(16))) __bf16 v16bf;
typedef __attribute__((ext_vector_type(8)))  float  v8f;

#define NP 100000
#define NA 50000
#define NPPAD 100032   // round up to 64-row tiles (async tile reads stay in-buffer)
#define NAPAD 50048
#define EE 250000

__device__ inline unsigned short f2bf(float f) {
  unsigned int x = __float_as_uint(f);
  unsigned int r = (x + 0x7FFFu + ((x >> 16) & 1u)) >> 16;   // round-to-nearest-even
  return (unsigned short)r;
}

// order-preserving float<->uint encoding for atomicMax on signed floats
__device__ inline unsigned fenc(float f) {
  unsigned u = __float_as_uint(f);
  return (u & 0x80000000u) ? ~u : (u | 0x80000000u);
}
__device__ inline float fdec(unsigned u) {
  u = (u & 0x80000000u) ? (u & 0x7fffffffu) : ~u;
  return __uint_as_float(u);
}

// CDNA5 async copy: 16B global -> LDS, tracked by ASYNCcnt
__device__ inline void async_g2l_b128(unsigned lds_off, const void* gptr) {
  asm volatile("global_load_async_to_lds_b128 %0, %1, off"
               :: "v"(lds_off), "v"(gptr) : "memory");
}
__device__ inline void wait_async0() {
  asm volatile("s_wait_asynccnt 0" ::: "memory");
}

// ---------------------------------------------------------------------------
// C[M,128] = Abf[M,128](bf16) @ W(bf16, pre-transposed Wt[col][k]) + bias
// optional epilogue: C = g*C + (1-g)*Xskip,  g = sigmoid(*skipPtr)
// Block: 256 threads = 8 waves; tile 64 rows x 128 cols; each wave: 16x64.
// ---------------------------------------------------------------------------
__global__ __launch_bounds__(256) void hgt_gemm128_wmma(
    const unsigned short* __restrict__ Abf, const unsigned short* __restrict__ Wt,
    const float* __restrict__ bias, const float* __restrict__ Xskip,
    const float* __restrict__ skipPtr, float* __restrict__ C, int M)
{
  __shared__ __align__(16) unsigned short lA[64][32];    // 64 rows  x 32 K (bf16)
  __shared__ __align__(16) unsigned short lB[128][32];   // 128 cols x 32 K (bf16)

  const int tid  = threadIdx.x;
  const int lane = tid & 31;
  const int wave = tid >> 5;
  const int wr   = wave >> 1;          // 0..3 : 16-row strip
  const int wc   = wave & 1;           // 0..1 : 64-col half
  const int r16  = lane & 15;
  const int hi   = lane >> 4;
  const int rowBase = blockIdx.x * 64;

  const unsigned ldsA = (unsigned)(size_t)&lA[0][0];
  const unsigned ldsB = (unsigned)(size_t)&lB[0][0];

  v8f acc[4];
  #pragma unroll
  for (int i = 0; i < 4; ++i)
    #pragma unroll
    for (int j = 0; j < 8; ++j) acc[i][j] = 0.0f;

  for (int kk = 0; kk < 4; ++kk) {
    if (kk < 3) __builtin_prefetch((const char*)Wt + (kk + 1) * 64 + (size_t)tid * 256, 0, 0);

    // ---- async stage A tile: 64 rows x 64B slab (256 x 16B chunks) ----
    {
      int rA = tid >> 2, pA = (tid & 3) * 16;
      async_g2l_b128(ldsA + rA * 64 + pA,
                     (const char*)Abf + (size_t)(rowBase + rA) * 256 + kk * 64 + pA);
    }
    // ---- async stage B tile: 128 cols x 64B slab (512 x 16B chunks) ----
    #pragma unroll
    for (int i = 0; i < 2; ++i) {
      int c = tid * 2 + i;
      int col = c >> 2, p = (c & 3) * 16;
      async_g2l_b128(ldsB + col * 64 + p,
                     (const char*)Wt + (size_t)col * 256 + kk * 64 + p);
    }
    wait_async0();
    __syncthreads();

    // ---- fragments (documented gfx1250 bf16 layouts) ----
    union FragU { v16bf v; uint4 q[2]; } afr;
    const int arow = wr * 16 + r16;
    afr.q[0] = *(const uint4*)&lA[arow][8 * hi];          // e 0..7  : K = e + 8*hi
    afr.q[1] = *(const uint4*)&lA[arow][16 + 8 * hi];     // e 8..15 : K = 16 + (e-8) + 8*hi
    #pragma unroll
    for (int ct = 0; ct < 4; ++ct) {
      union FragU bfr;
      const int bcol = wc * 64 + ct * 16 + r16;
      bfr.q[0] = *(const uint4*)&lB[bcol][8 * hi];        // e 0..7 : K = e + 8*hi; e 8..15 : K = 16 + (e-8) + 8*hi
      bfr.q[1] = *(const uint4*)&lB[bcol][16 + 8 * hi];
      acc[ct] = __builtin_amdgcn_wmma_f32_16x16x32_bf16(
          false, afr.v, false, bfr.v, (short)0, acc[ct], false, false);
    }
    __syncthreads();
  }

  // ---- epilogue: bias (+ skip-gate blend) ----
  float g = 1.0f, gi = 0.0f;
  if (skipPtr) { float sv = *skipPtr; g = 1.0f / (1.0f + __expf(-sv)); gi = 1.0f - g; }
  #pragma unroll
  for (int ct = 0; ct < 4; ++ct) {
    const int col = wc * 64 + ct * 16 + r16;
    const float b = bias ? bias[col] : 0.0f;
    #pragma unroll
    for (int j = 0; j < 8; ++j) {
      const int row = rowBase + wr * 16 + j + 8 * hi;     // C/D: M = j + 8*hi
      if (row < M) {
        float val = acc[ct][j] + b;
        if (Xskip) val = g * val + gi * Xskip[(size_t)row * 128 + col];
        C[(size_t)row * 128 + col] = val;
      }
    }
  }
}

// f32 -> bf16 activation conversion, fused scale + exact GELU
__global__ void hgt_cvt_act(const float* __restrict__ src, unsigned short* __restrict__ dst,
                            int n, float scale, int doGelu)
{
  int i = blockIdx.x * blockDim.x + threadIdx.x;
  if (i >= n) return;
  float x = src[i] * scale;
  if (doGelu) x = 0.5f * x * (1.0f + erff(x * 0.70710678118f));
  dst[i] = f2bf(x);
}

// f32 weight [k][c] -> bf16 transposed Wt[c][k]
__global__ void hgt_wt_cvt(const float* __restrict__ W, unsigned short* __restrict__ Wt)
{
  int i = blockIdx.x * blockDim.x + threadIdx.x;          // 16384
  int k = i >> 7, c = i & 127;
  Wt[(size_t)c * 128 + k] = f2bf(W[i]);
}

// ---------------------------------------------------------------------------
// Fold relation transform into projection weights, emit transposed bf16:
//   WcT[c][r] = bf16( sum_d src_row(r)[h*32+d] * a_rel[h][d][j] ),  c = h*32+j
//   row 128 = bias (kept f32)
// ---------------------------------------------------------------------------
__global__ void hgt_combine_w(const float* __restrict__ w, const float* __restrict__ b,
                              const float* __restrict__ arel,
                              unsigned short* __restrict__ wcT, float* __restrict__ bc)
{
  const int r = blockIdx.x;            // 0..128 (128 == bias row)
  const int c = threadIdx.x;           // 0..127
  const int h = c >> 5, j = c & 31;
  const float* ap  = arel + h * 1024 + j;                      // stride 32 over d
  const float* src = (r < 128) ? (w + (size_t)r * 128 + h * 32) : (b + h * 32);
  float acc = 0.0f;
  #pragma unroll
  for (int d = 0; d < 32; ++d) acc += src[d] * ap[d * 32];
  if (r < 128) wcT[(size_t)c * 128 + r] = f2bf(acc); else bc[c] = acc;
}

__global__ void hgt_gather_rows(float* __restrict__ dst, const float* __restrict__ table,
                                const int* __restrict__ idx, int n)
{
  int i = blockIdx.x * blockDim.x + threadIdx.x;
  if (i >= n * 128) return;
  int r = i >> 7, c = i & 127;
  dst[i] = table[(size_t)idx[r] * 128 + c];
}

// pass A: attention scores + per-(dst,head) running max (encoded atomicMax)
__global__ void hgt_edge_scores(const float* __restrict__ q, const float* __restrict__ krel,
                                const int* __restrict__ src, const int* __restrict__ dst,
                                const float* __restrict__ prel,
                                float* __restrict__ score, unsigned* __restrict__ maxb)
{
  int i = blockIdx.x * blockDim.x + threadIdx.x;
  if (i >= EE * 4) return;
  int e = i >> 2, h = i & 3;
  int s = src[e], d = dst[e];
  const float* qp = q    + (size_t)d * 128 + h * 32;
  const float* kp = krel + (size_t)s * 128 + h * 32;
  float acc = 0.0f;
  #pragma unroll
  for (int j = 0; j < 32; ++j) acc += qp[j] * kp[j];
  acc *= prel[h] * 0.17677669529663687f;                       // 1/sqrt(32)
  score[i] = acc;
  atomicMax(maxb + (size_t)d * 4 + h, fenc(acc));
}

// pass B: exp(score - max) and denominator accumulate
__global__ void hgt_edge_norm(const int* __restrict__ dst, const unsigned* __restrict__ maxb,
                              float* __restrict__ score, float* __restrict__ denb)
{
  int i = blockIdx.x * blockDim.x + threadIdx.x;
  if (i >= EE * 4) return;
  int e = i >> 2, h = i & 3;
  int d = dst[e];
  float m  = fdec(maxb[(size_t)d * 4 + h]);
  float ex = __expf(score[i] - m);
  score[i] = ex;
  atomicAdd(denb + (size_t)d * 4 + h, ex);
}

// pass C: weighted message scatter-add
__global__ void hgt_edge_scatter(const float* __restrict__ vrel, const int* __restrict__ src,
                                 const int* __restrict__ dst, const float* __restrict__ score,
                                 const float* __restrict__ denb, float* __restrict__ out)
{
  int i = blockIdx.x * blockDim.x + threadIdx.x;
  if (i >= EE * 4) return;
  int e = i >> 2, h = i & 3;
  int s = src[e], d = dst[e];
  float w = score[i] / (denb[(size_t)d * 4 + h] + 1e-16f);
  const float* vp = vrel + (size_t)s * 128 + h * 32;
  float*       op = out  + (size_t)d * 128 + h * 32;
  #pragma unroll
  for (int j = 0; j < 32; ++j) atomicAdd(op + j, vp[j] * w);
}

__global__ void hgt_copy(float* __restrict__ dst, const float* __restrict__ src, int n)
{
  int i = blockIdx.x * blockDim.x + threadIdx.x;
  if (i < n) dst[i] = src[i];
}

// ---------------------------------------------------------------------------
extern "C" void kernel_launch(void* const* d_in, const int* in_sizes, int n_in,
                              void* d_out, int out_size, void* d_ws, size_t ws_size,
                              hipStream_t stream)
{
  (void)in_sizes; (void)n_in; (void)out_size; (void)ws_size;

  const float* x_paper = (const float*)d_in[0];
  const int*   a_idx   = (const int*)  d_in[1];
  const float* embed   = (const float*)d_in[2];
  const float* lin_w   = (const float*)d_in[3];
  const float* lin_b   = (const float*)d_in[4];
  const float* k_w     = (const float*)d_in[5];
  const float* k_b     = (const float*)d_in[6];
  const float* q_w     = (const float*)d_in[7];
  const float* q_b     = (const float*)d_in[8];
  const float* v_w     = (const float*)d_in[9];
  const float* v_b     = (const float*)d_in[10];
  const float* a_rel   = (const float*)d_in[11];
  const float* m_rel   = (const float*)d_in[12];
  const float* p_rel   = (const float*)d_in[13];
  const float* skip    = (const float*)d_in[14];
  const float* a_w     = (const float*)d_in[15];
  const float* a_b     = (const float*)d_in[16];
  const int* esrc[3] = {(const int*)d_in[17], (const int*)d_in[19], (const int*)d_in[21]};
  const int* edst[3] = {(const int*)d_in[18], (const int*)d_in[20], (const int*)d_in[22]};

  // workspace carve-up (float units)
  float* ws = (float*)d_ws;
  size_t off = 0;
  auto alloc = [&](size_t n) { float* p = ws + off; off += n; return p; };
  float* xs[2];   xs[0]  = alloc((size_t)NP * 128); xs[1]  = alloc((size_t)NA * 128);
  float* nx[2];   nx[0]  = alloc((size_t)NP * 128); nx[1]  = alloc((size_t)NA * 128);
  float* qb[2];   qb[0]  = alloc((size_t)NP * 128); qb[1]  = alloc((size_t)NA * 128);
  float* krel   = alloc((size_t)NP * 128);
  float* vrel   = alloc((size_t)NP * 128);
  float* outb[2]; outb[0] = alloc((size_t)NP * 128); outb[1] = alloc((size_t)NA * 128);
  unsigned* maxb = (unsigned*)alloc((size_t)NP * 4);
  float* denb   = alloc((size_t)NP * 4);
  float* score  = alloc((size_t)EE * 4);
  float* bkC = alloc(128);
  float* bvC = alloc(128);
  // bf16 buffers (2 bytes/elem -> halves of float units)
  unsigned short* xpb    = (unsigned short*)alloc((size_t)NPPAD * 64);
  unsigned short* xsb[2];
  xsb[0] = (unsigned short*)alloc((size_t)NPPAD * 64);
  xsb[1] = (unsigned short*)alloc((size_t)NAPAD * 64);
  unsigned short* obb    = (unsigned short*)alloc((size_t)NPPAD * 64);
  unsigned short* Wt0    = (unsigned short*)alloc(8192);
  unsigned short* WkT    = (unsigned short*)alloc(8192);
  unsigned short* WvT    = (unsigned short*)alloc(8192);

  auto gemm = [&](const unsigned short* A, const unsigned short* Wt, const float* bias,
                  const float* Xs, const float* sp, float* Cc, int M) {
    hgt_gemm128_wmma<<<dim3((M + 63) / 64), dim3(256), 0, stream>>>(A, Wt, bias, Xs, sp, Cc, M);
  };
  auto cvt = [&](const float* s, unsigned short* d, int n, float sc, int gl) {
    hgt_cvt_act<<<(n + 255) / 256, 256, 0, stream>>>(s, d, n, sc, gl);
  };
  auto wtcvt = [&](const float* w, unsigned short* wt) {
    hgt_wt_cvt<<<64, 256, 0, stream>>>(w, wt);
  };

  const int Ns[2] = {NP, NA};
  const int ETY[3][2] = {{1, 0}, {0, 1}, {0, 0}};   // (src type, dst type)
  const int gE = (EE * 4 + 255) / 256;

  // input projections
  cvt(x_paper, xpb, NP * 128, 1.0f, 0);
  wtcvt(lin_w, Wt0);
  gemm(xpb, Wt0, lin_b, nullptr, nullptr, xs[0], NP);
  hgt_gather_rows<<<(NA * 128 + 255) / 256, 256, 0, stream>>>(xs[1], embed, a_idx, NA);

  for (int l = 0; l < 2; ++l) {
    cvt(xs[0], xsb[0], NP * 128, 1.0f, 0);
    cvt(xs[1], xsb[1], NA * 128, 1.0f, 0);

    for (int t = 0; t < 2; ++t) {
      wtcvt(q_w + (size_t)(l * 2 + t) * 16384, Wt0);
      gemm(xsb[t], Wt0, q_b + (size_t)(l * 2 + t) * 128, nullptr, nullptr, qb[t], Ns[t]);
    }

    hipMemsetAsync(outb[0], 0, (size_t)NP * 128 * sizeof(float), stream);
    hipMemsetAsync(outb[1], 0, (size_t)NA * 128 * sizeof(float), stream);

    for (int e = 0; e < 3; ++e) {
      const int s = ETY[e][0], d = ETY[e][1];
      const float* ar = a_rel + (size_t)(l * 3 + e) * 4096;
      const float* mr = m_rel + (size_t)(l * 3 + e) * 4096;
      hgt_combine_w<<<129, 128, 0, stream>>>(k_w + (size_t)(l * 2 + s) * 16384,
                                             k_b + (size_t)(l * 2 + s) * 128, ar, WkT, bkC);
      hgt_combine_w<<<129, 128, 0, stream>>>(v_w + (size_t)(l * 2 + s) * 16384,
                                             v_b + (size_t)(l * 2 + s) * 128, mr, WvT, bvC);
      gemm(xsb[s], WkT, bkC, nullptr, nullptr, krel, Ns[s]);
      gemm(xsb[s], WvT, bvC, nullptr, nullptr, vrel, Ns[s]);

      hipMemsetAsync(maxb, 0, (size_t)Ns[d] * 4 * sizeof(unsigned), stream);
      hipMemsetAsync(denb, 0, (size_t)Ns[d] * 4 * sizeof(float), stream);

      hgt_edge_scores<<<gE, 256, 0, stream>>>(qb[d], krel, esrc[e], edst[e],
                                              p_rel + (size_t)(l * 3 + e) * 4, score, maxb);
      hgt_edge_norm<<<gE, 256, 0, stream>>>(edst[e], maxb, score, denb);
      hgt_edge_scatter<<<gE, 256, 0, stream>>>(vrel, esrc[e], edst[e], score, denb, outb[d]);
    }

    const float cnt[2] = {2.0f, 1.0f};   // paper dst: writes+cites; author dst: rev_writes
    for (int t = 0; t < 2; ++t) {
      cvt(outb[t], obb, Ns[t] * 128, 1.0f / cnt[t], 1);       // mean + exact GELU, bf16
      wtcvt(a_w + (size_t)(l * 2 + t) * 16384, Wt0);
      gemm(obb, Wt0, a_b + (size_t)(l * 2 + t) * 128, xs[t], skip + (size_t)(l * 2 + t),
           nx[t], Ns[t]);
    }

    float* tp;
    tp = xs[0]; xs[0] = nx[0]; nx[0] = tp;
    tp = xs[1]; xs[1] = nx[1]; nx[1] = tp;
  }

  hgt_copy<<<(50000 * 128 + 255) / 256, 256, 0, stream>>>((float*)d_out, xs[0], 50000 * 128);
  hgt_copy<<<(25000 * 128 + 255) / 256, 256, 0, stream>>>((float*)d_out + 50000 * 128, xs[1],
                                                          25000 * 128);
}
